// BinaryTreeLSTM_36661840839776
// MI455X (gfx1250) — compile-verified, matrix operand order checked by reading, not run on hardware
//
#include <hip/hip_runtime.h>
#include <math.h>

// ---------------------------------------------------------------------------
// BinaryTreeLSTM on MI455X (gfx1250, wave32, WMMA).
//   Per level k (n=2^k rows): gates(n x 4096) = [x|h_cat](n x 3072) @ Wc^T + b
//   (only cols 0..1023 of each gate are live), fused LSTM cell update.
// v_wmma_f32_16x16x32_bf16, 2x2 register-blocked 32x32 tile per wave:
//   2 A-frags + 2 B-frags -> 4 WMMAs per K-step (2 b128 loads per WMMA),
//   halving L2 traffic on the L2-resident 24MB weight matrix.
// ---------------------------------------------------------------------------

typedef __bf16 bf16_t;
typedef __attribute__((ext_vector_type(16))) __bf16 v16bf;
typedef __attribute__((ext_vector_type(8)))  float  v8f;

#define TDEPTH 12
#define HSZ    1024
#define KTOT   3072   // 1024 (x) + 2048 (h_cat)

// ---- one-time: compact weights to bf16, fold bias, zero initial h/c -------
__global__ void tree_lstm_init(const float* __restrict__ W_ih,
                               const float* __restrict__ W_hh,
                               const float* __restrict__ b_ih,
                               const float* __restrict__ b_hh,
                               bf16_t* __restrict__ Wc,
                               float*  __restrict__ bc,
                               float*  __restrict__ h0,
                               float*  __restrict__ c0)
{
    long idx = (long)blockIdx.x * blockDim.x + threadIdx.x;
    const long WCE = 4096L * KTOT;
    if (idx < WCE) {
        int q   = (int)(idx / KTOT);     // compacted gate row: gate*1024 + col
        int k   = (int)(idx % KTOT);
        int j   = q >> 10;               // gate 0..3
        int col = q & 1023;
        int src = j * 2048 + col;        // original row inside 8H
        float v = (k < 1024) ? W_ih[(long)src * 1024 + k]
                             : W_hh[(long)src * 2048 + (k - 1024)];
        Wc[idx] = (bf16_t)v;
    }
    if (idx < 4096) {
        int j = (int)(idx >> 10), col = (int)(idx & 1023);
        int src = j * 2048 + col;
        bc[idx] = b_ih[src] + b_hh[src];
    }
    if (idx < 4096L * HSZ) {
        h0[idx] = 0.0f;
        c0[idx] = 0.0f;
    }
}

// ---- per level: build bf16 A = [x | h_cat], zero-pad to 32-row multiple ---
__global__ void tree_lstm_prep(const float* __restrict__ emb,
                               const float* __restrict__ h_in,
                               bf16_t* __restrict__ A,
                               int n, int Mp)
{
    long idx = (long)blockIdx.x * blockDim.x + threadIdx.x;
    long tot = (long)Mp * KTOT;
    if (idx >= tot) return;
    int r = (int)(idx / KTOT);
    int k = (int)(idx % KTOT);
    float v = 0.0f;
    if (r < n) {
        v = (k < 1024) ? emb[(long)(n - 1 + r) * 1024 + k]
                       : h_in[(long)r * 2048 + (k - 1024)];   // h.reshape(n,2H)
    }
    A[idx] = (bf16_t)v;
}

// ---- per level: 4-wave block, wave j = gate j, 32x32 tile per wave via
// 2x2 WMMA register blocking. Gates meet in LDS; fused cell update. ---------
__global__ __launch_bounds__(128)
void tree_lstm_level(const bf16_t* __restrict__ A,
                     const bf16_t* __restrict__ Wc,
                     const float*  __restrict__ bc,
                     const float*  __restrict__ c_in,
                     float* __restrict__ h_out,
                     float* __restrict__ c_out,
                     int nrows)
{
    __shared__ float glds[4][32][32];   // [gate][m][n] 16 KB

    const int tid  = threadIdx.x;
    const int j    = tid >> 5;          // wave id == gate id (i,f,g,o)
    const int lane = tid & 31;
    const int nl   = lane & 15;         // M index (A frag) / N index (B frag)
    const int kh   = lane >> 4;         // K half-select within fragment
    const int m0   = blockIdx.x << 5;   // 32-row tile
    const int c0   = blockIdx.y << 5;   // 32-col tile (0..992)

    // Per-lane contiguous 32B fragments (row-major A; B[k,n] = Wc[q0+n, k]).
    const bf16_t* Ap0 = A  + (long)(m0 + nl) * KTOT + kh * 16;
    const bf16_t* Ap1 = Ap0 + 16L * KTOT;
    const bf16_t* Bp0 = Wc + (long)(j * 1024 + c0 + nl) * KTOT + kh * 16;
    const bf16_t* Bp1 = Bp0 + 16L * KTOT;

    v8f acc00 = {}, acc01 = {}, acc10 = {}, acc11 = {};
    #pragma unroll 2
    for (int k = 0; k < KTOT; k += 32) {
        v16bf a0 = *(const v16bf*)(Ap0 + k);
        v16bf a1 = *(const v16bf*)(Ap1 + k);
        v16bf b0 = *(const v16bf*)(Bp0 + k);
        v16bf b1 = *(const v16bf*)(Bp1 + k);
        acc00 = __builtin_amdgcn_wmma_f32_16x16x32_bf16(false, a0, false, b0,
                                                        (short)0, acc00, false, false);
        acc01 = __builtin_amdgcn_wmma_f32_16x16x32_bf16(false, a0, false, b1,
                                                        (short)0, acc01, false, false);
        acc10 = __builtin_amdgcn_wmma_f32_16x16x32_bf16(false, a1, false, b0,
                                                        (short)0, acc10, false, false);
        acc11 = __builtin_amdgcn_wmma_f32_16x16x32_bf16(false, a1, false, b1,
                                                        (short)0, acc11, false, false);
    }

    // C/D layout: VGPR v, lanes 0-15 -> M=v, lanes 16-31 -> M=8+v; N=lane&15.
    const float bias0 = bc[j * 1024 + c0 + nl];
    const float bias1 = bc[j * 1024 + c0 + 16 + nl];
    const int   mrow  = 8 * kh;
    #pragma unroll
    for (int v = 0; v < 8; ++v) {
        glds[j][     mrow + v][     nl] = acc00[v] + bias0;
        glds[j][     mrow + v][16 + nl] = acc01[v] + bias1;
        glds[j][16 + mrow + v][     nl] = acc10[v] + bias0;
        glds[j][16 + mrow + v][16 + nl] = acc11[v] + bias1;
    }
    __syncthreads();

    #pragma unroll
    for (int e = tid; e < 1024; e += 128) {
        int m = e >> 5, nn = e & 31;
        int row = m0 + m, col = c0 + nn;
        if (row < nrows) {
            float iv = glds[0][m][nn];
            float fv = glds[1][m][nn];
            float gv = glds[2][m][nn];
            float ov = glds[3][m][nn];
            float ccat = c_in[(long)row * 2048 + col];   // c.reshape(n,2H)
            float si = 1.0f / (1.0f + __expf(-iv));
            float sf = 1.0f / (1.0f + __expf(-fv));
            float so = 1.0f / (1.0f + __expf(-ov));
            float cn = sf * ccat + si * tanhf(gv);
            float hn = so * tanhf(cn);
            h_out[(long)row * HSZ + col] = hn;
            c_out[(long)row * HSZ + col] = cn;
        }
    }
}

extern "C" void kernel_launch(void* const* d_in, const int* in_sizes, int n_in,
                              void* d_out, int out_size, void* d_ws, size_t ws_size,
                              hipStream_t stream)
{
    (void)in_sizes; (void)n_in; (void)out_size; (void)ws_size;
    const float* emb  = (const float*)d_in[0];
    const float* W_ih = (const float*)d_in[1];
    const float* W_hh = (const float*)d_in[2];
    const float* b_ih = (const float*)d_in[3];
    const float* b_hh = (const float*)d_in[4];
    float* out = (float*)d_out;

    // Workspace carve-up (~105 MB), 256B-aligned slices.
    char* ws = (char*)d_ws;
    size_t off = 0;
    auto carve = [&](size_t bytes) -> void* {
        void* p = ws + off;
        off = (off + bytes + 255) & ~(size_t)255;
        return p;
    };
    bf16_t* Wc   = (bf16_t*)carve(4096UL * KTOT * sizeof(bf16_t));  // 24 MB
    float*  bc   = (float*) carve(4096UL * sizeof(float));
    float*  hA   = (float*) carve(4096UL * HSZ * sizeof(float));    // 16 MB
    float*  hB   = (float*) carve(4096UL * HSZ * sizeof(float));
    float*  cA   = (float*) carve(4096UL * HSZ * sizeof(float));
    float*  cB   = (float*) carve(4096UL * HSZ * sizeof(float));
    bf16_t* Abuf = (bf16_t*)carve(2048UL * KTOT * sizeof(bf16_t));  // 12.6 MB

    {
        long tot = 4096L * KTOT;
        int blocks = (int)((tot + 255) / 256);
        tree_lstm_init<<<blocks, 256, 0, stream>>>(W_ih, W_hh, b_ih, b_hh,
                                                   Wc, bc, hA, cA);
    }

    const float* h_in = hA;  const float* c_in = cA;
    float*       h_wr = hB;  float*       c_wr = cB;

    for (int k = TDEPTH - 1; k >= 0; --k) {
        int n  = 1 << k;
        int Mp = (n + 31) & ~31;            // pad to 32-row tiles

        long tot = (long)Mp * KTOT;
        tree_lstm_prep<<<(int)((tot + 255) / 256), 256, 0, stream>>>(
            emb, h_in, Abuf, n, Mp);

        float* ho = (k == 0) ? out       : h_wr;   // final level writes [h|c]
        float* co = (k == 0) ? out + HSZ : c_wr;   // straight into d_out
        dim3 grid(Mp / 32, HSZ / 32);              // 32x32 tiles
        tree_lstm_level<<<grid, 128, 0, stream>>>(Abuf, Wc, bc, c_in,
                                                  ho, co, n);

        const float* th = h_in; h_in = h_wr; h_wr = (float*)th;
        const float* tc = c_in; c_in = c_wr; c_wr = (float*)tc;
    }
}
